// TransformerModel_21380347200242
// MI455X (gfx1250) — matrix operand order repaired and checked by reading, not compiled, not run on hardware
//
#include <hip/hip_runtime.h>

typedef long long ll;
typedef __attribute__((ext_vector_type(16))) __bf16 v16bf;
typedef __attribute__((ext_vector_type(8)))  float  v8f;

typedef union { unsigned int u[8]; v16bf v; } frag_t;

// ---------- helpers ----------
__device__ __forceinline__ unsigned short f2bf(float f) {
    unsigned int x = __float_as_uint(f);
    unsigned int r = x + 0x7FFFu + ((x >> 16) & 1u);   // RNE
    return (unsigned short)(r >> 16);
}

// ---------- generic batched WMMA GEMM (bf16 in, fp32 accum) ----------
// Y[z][m][n] = act( scale * sum_k A[z][m*lda + k] * B[z][n*ldb + k] + bias[n] )
// batch offset for X in {A,B,Y}: off = (z/HB)*sB + (z%HB)*sH
// Requires: M%64==0 (grid.y), N%64==0 (grid.x), K%64==0, all row-contiguous in K.
__global__ __launch_bounds__(128)
void wmma_gemm_bf16(const unsigned short* __restrict__ A, ll saB, ll saH, int lda,
                    const unsigned short* __restrict__ B, ll sbB, ll sbH, int ldb,
                    float* __restrict__ Yf, unsigned short* __restrict__ Yh,
                    ll syB, ll syH, int ldy,
                    const float* __restrict__ bias, float scale, int K, int act, int HB)
{
    int z  = blockIdx.z;
    int zb = z / HB, zh = z % HB;
    const unsigned short* Ab = A + zb * saB + zh * saH;
    const unsigned short* Bb = B + zb * sbB + zh * sbH;
    ll yoff = (ll)zb * syB + (ll)zh * syH;

    int wave = threadIdx.x >> 5;
    int lane = threadIdx.x & 31;
    int hi   = lane >> 4;      // lane half
    int l15  = lane & 15;
    int m_base = blockIdx.y * 64 + wave * 16;
    int n_base = blockIdx.x * 64;

    // A-fragment K offsets (16x32 bf16 layout, ISA 7.12.2):
    // lanes 0-15 hold K {0..7,16..23}, lanes 16-31 hold K {8..15,24..31}
    const unsigned short* Arow = Ab + (ll)(m_base + l15) * lda + hi * 8;
    // B-fragment: same K layout as A (lanes 0-15 K {0..7,16..23}, lanes 16-31 K {8..15,24..31})
    const unsigned short* Brow0 = Bb + (ll)(n_base + 0 * 16 + l15) * ldb + hi * 8;
    const unsigned short* Brow1 = Bb + (ll)(n_base + 1 * 16 + l15) * ldb + hi * 8;
    const unsigned short* Brow2 = Bb + (ll)(n_base + 2 * 16 + l15) * ldb + hi * 8;
    const unsigned short* Brow3 = Bb + (ll)(n_base + 3 * 16 + l15) * ldb + hi * 8;

    v8f acc[4];
#pragma unroll
    for (int j = 0; j < 4; ++j) { v8f zinit = {}; acc[j] = zinit; }

#pragma unroll 2
    for (int k0 = 0; k0 < K; k0 += 32) {
        // Issue ALL loads of this K-step first (distinct regs) so the compiler
        // can overlap trailing loads with leading WMMAs via partial waits.
        frag_t a, b[4];
#pragma unroll
        for (int r = 0; r < 4; ++r) {
            a.u[r]     = *(const unsigned int*)(Arow + k0 + 2 * r);        // K 0..7  (+hi*8)
            a.u[r + 4] = *(const unsigned int*)(Arow + k0 + 16 + 2 * r);   // K 16..23(+hi*8)
        }
        const unsigned short* brows[4] = { Brow0, Brow1, Brow2, Brow3 };
#pragma unroll
        for (int j = 0; j < 4; ++j)
#pragma unroll
            for (int r = 0; r < 8; ++r)
                b[j].u[r] = *(const unsigned int*)(brows[j] + k0 + 2 * r + ((r >> 2) << 3));

#pragma unroll
        for (int j = 0; j < 4; ++j)
            acc[j] = __builtin_amdgcn_wmma_f32_16x16x32_bf16(
                false, a.v, false, b[j].v, (short)0, acc[j], false, false);
    }

#pragma unroll
    for (int j = 0; j < 4; ++j) {
        int col = n_base + j * 16 + l15;
        float bv = bias ? bias[col] : 0.0f;
#pragma unroll
        for (int r = 0; r < 8; ++r) {
            int row = m_base + r + hi * 8;
            float val = acc[j][r] * scale + bv;
            if (act == 1)  // exact GELU (erf variant)
                val = 0.5f * val * (1.0f + erff(val * 0.7071067811865475f));
            ll idx = yoff + (ll)row * ldy + col;
            if (Yf) Yf[idx] = val;
            if (Yh) Yh[idx] = f2bf(val);
        }
    }
}

// ---------- V transpose: [B,S,D] bf16 -> [B,H,DH,S] bf16 ----------
__global__ void transpose_v_kernel(const unsigned short* __restrict__ vin,
                                   unsigned short* __restrict__ vout,
                                   int S, int D, int H, int DH, int n)
{
    int i = blockIdx.x * blockDim.x + threadIdx.x;   // over B*H*DH*S
    if (i >= n) return;
    int k  = i % S;
    int t  = i / S;
    int d  = t % DH;
    int bh = t / DH;          // b*H + h
    int b  = bh / H;
    int h  = bh % H;
    vout[i] = vin[((ll)b * S + k) * D + h * DH + d];
}

// ---------- embedding: tok_emb[tok]*sqrt(D) + pe[s] ----------
__global__ void embed_kernel(const int* __restrict__ tok, const float* __restrict__ emb,
                             const float* __restrict__ pe, float* __restrict__ out,
                             int S, int D, float sqrtD)
{
    int row = blockIdx.x;            // b*S + s
    int s = row % S;
    int t = tok[row];
    const float* e = emb + (ll)t * D;
    const float* p = pe  + (ll)s * D;
    float* o = out + (ll)row * D;
    for (int d = threadIdx.x; d < D; d += blockDim.x)
        o[d] = e[d] * sqrtD + p[d];
}

// ---------- LayerNorm (fp32 in -> bf16 out) ----------
__global__ __launch_bounds__(256)
void ln_bf16_kernel(const float* __restrict__ x, const float* __restrict__ g,
                    const float* __restrict__ b, unsigned short* __restrict__ out,
                    int D, float eps)
{
    int row = blockIdx.x;
    const float* xr = x + (ll)row * D;
    __shared__ float red[256];
    float s = 0.f;
    for (int d = threadIdx.x; d < D; d += 256) s += xr[d];
    red[threadIdx.x] = s; __syncthreads();
    for (int off = 128; off > 0; off >>= 1) {
        if (threadIdx.x < off) red[threadIdx.x] += red[threadIdx.x + off];
        __syncthreads();
    }
    float mean = red[0] / D; __syncthreads();
    float v = 0.f;
    for (int d = threadIdx.x; d < D; d += 256) { float t = xr[d] - mean; v += t * t; }
    red[threadIdx.x] = v; __syncthreads();
    for (int off = 128; off > 0; off >>= 1) {
        if (threadIdx.x < off) red[threadIdx.x] += red[threadIdx.x + off];
        __syncthreads();
    }
    float rstd = rsqrtf(red[0] / D + eps);
    unsigned short* orow = out + (ll)row * D;
    for (int d = threadIdx.x; d < D; d += 256)
        orow[d] = f2bf((xr[d] - mean) * rstd * g[d] + b[d]);
}

// ---------- masked softmax over a score row (fp32 in -> bf16 probs) ----------
__global__ __launch_bounds__(256)
void softmax_bf16(const float* __restrict__ scores, unsigned short* __restrict__ probs,
                  const unsigned char* __restrict__ kpm, int S, int H, int causal)
{
    int q  = blockIdx.x;
    int bh = blockIdx.y;
    int b  = bh / H;
    const float* srow = scores + ((ll)bh * S + q) * (ll)S;
    unsigned short* prow = probs + ((ll)bh * S + q) * (ll)S;
    const unsigned char* kp = kpm + (ll)b * S;
    __shared__ float red[256];

    const int PER = 4;               // S==1024, 256 threads
    float vals[PER];
    float mx = -3.4e38f;
#pragma unroll
    for (int i = 0; i < PER; ++i) {
        int kpos = threadIdx.x + i * 256;
        float v = srow[kpos];
        bool masked = (kp[kpos] != 0) || (causal && (kpos > q));
        v = masked ? -3.4e38f : v;
        vals[i] = v;
        mx = fmaxf(mx, v);
    }
    red[threadIdx.x] = mx; __syncthreads();
    for (int off = 128; off > 0; off >>= 1) {
        if (threadIdx.x < off) red[threadIdx.x] = fmaxf(red[threadIdx.x], red[threadIdx.x + off]);
        __syncthreads();
    }
    mx = red[0]; __syncthreads();
    float sum = 0.f;
#pragma unroll
    for (int i = 0; i < PER; ++i) {
        float e = (vals[i] <= -3.3e38f) ? 0.f : expf(vals[i] - mx);
        vals[i] = e; sum += e;
    }
    red[threadIdx.x] = sum; __syncthreads();
    for (int off = 128; off > 0; off >>= 1) {
        if (threadIdx.x < off) red[threadIdx.x] += red[threadIdx.x + off];
        __syncthreads();
    }
    float inv = 1.0f / red[0];
#pragma unroll
    for (int i = 0; i < PER; ++i) {
        int kpos = threadIdx.x + i * 256;
        prow[kpos] = f2bf(vals[i] * inv);
    }
}

// ---------- elementwise ----------
__global__ void add_kernel(float* __restrict__ x, const float* __restrict__ y, int n)
{
    int i = blockIdx.x * blockDim.x + threadIdx.x;
    if (i < n) x[i] += y[i];
}
__global__ void cvt_f32_bf16(const float* __restrict__ in, unsigned short* __restrict__ out, int n)
{
    int i = blockIdx.x * blockDim.x + threadIdx.x;
    if (i < n) out[i] = f2bf(in[i]);
}

// =====================================================================
extern "C" void kernel_launch(void* const* d_in, const int* in_sizes, int n_in,
                              void* d_out, int out_size, void* d_ws, size_t ws_size,
                              hipStream_t stream)
{
    (void)in_sizes; (void)n_in; (void)out_size; (void)ws_size;
    const int Vv = 32000, Dd = 512, Hh = 8, Ff = 2048, Bb = 2, Ss = 1024, DH = 64;
    const int M = Bb * Ss;                  // 2048 rows
    const float SQRT_D = 22.62741699796952f;

    const int* src_tok = (const int*)d_in[0];
    const int* tgt_tok = (const int*)d_in[1];
    const unsigned char* skpm = (const unsigned char*)d_in[2];
    const unsigned char* tkpm = (const unsigned char*)d_in[3];
    auto F32 = [&](int i) { return (const float*)d_in[i]; };

    // ---- workspace bump allocator (256B aligned) ----
    char* wp = (char*)d_ws;
    size_t off = 0;
    auto alloc = [&](size_t bytes) -> void* {
        void* p = wp + off;
        off = (off + bytes + 255) & ~(size_t)255;
        return p;
    };
    auto cvtW = [&](const float* src, int n) -> unsigned short* {
        unsigned short* dst = (unsigned short*)alloc((size_t)n * 2);
        cvt_f32_bf16<<<(n + 255) / 256, 256, 0, stream>>>(src, dst, n);
        return dst;
    };

    struct Lin { unsigned short* w; const float* b; };
    struct MHA { Lin q, k, v, o; };
    struct EncL { MHA attn; Lin f1, f2; const float *n1g, *n1b, *n2g, *n2b; };
    struct DecL { MHA attn, xattn; Lin f1, f2;
                  const float *n1g, *n1b, *n2g, *n2b, *n3g, *n3b; };

    // pytree (sorted-key) flatten layout: dec[2] @4 (26 each), enc[2] @56 (16 each),
    // out.b=88, out.w=89, pe=90, tok_emb=91
    auto mkMHA = [&](int base) -> MHA {
        MHA m;
        m.k = { cvtW(F32(base + 1), Dd * Dd), F32(base + 0) };
        m.o = { cvtW(F32(base + 3), Dd * Dd), F32(base + 2) };
        m.q = { cvtW(F32(base + 5), Dd * Dd), F32(base + 4) };
        m.v = { cvtW(F32(base + 7), Dd * Dd), F32(base + 6) };
        return m;
    };
    DecL dec[2]; EncL enc[2];
    for (int l = 0; l < 2; ++l) {
        int b0 = 4 + l * 26;
        dec[l].attn = mkMHA(b0);
        dec[l].f1 = { cvtW(F32(b0 + 9),  Ff * Dd), F32(b0 + 8)  };
        dec[l].f2 = { cvtW(F32(b0 + 11), Dd * Ff), F32(b0 + 10) };
        dec[l].n1b = F32(b0 + 12); dec[l].n1g = F32(b0 + 13);
        dec[l].n2b = F32(b0 + 14); dec[l].n2g = F32(b0 + 15);
        dec[l].n3b = F32(b0 + 16); dec[l].n3g = F32(b0 + 17);
        dec[l].xattn = mkMHA(b0 + 18);
    }
    for (int l = 0; l < 2; ++l) {
        int b0 = 56 + l * 16;
        enc[l].attn = mkMHA(b0);
        enc[l].f1 = { cvtW(F32(b0 + 9),  Ff * Dd), F32(b0 + 8)  };
        enc[l].f2 = { cvtW(F32(b0 + 11), Dd * Ff), F32(b0 + 10) };
        enc[l].n1b = F32(b0 + 12); enc[l].n1g = F32(b0 + 13);
        enc[l].n2b = F32(b0 + 14); enc[l].n2g = F32(b0 + 15);
    }
    const float* out_b = F32(88);
    unsigned short* out_w = cvtW(F32(89), Vv * Dd);
    const float* pe      = F32(90);
    const float* tok_emb = F32(91);

    // ---- activation buffers ----
    float* xe  = (float*)alloc((size_t)M * Dd * 4);
    float* xd  = (float*)alloc((size_t)M * Dd * 4);
    unsigned short* memh = (unsigned short*)alloc((size_t)M * Dd * 2);
    unsigned short* hh   = (unsigned short*)alloc((size_t)M * Dd * 2);
    unsigned short* qh   = (unsigned short*)alloc((size_t)M * Dd * 2);
    unsigned short* kh   = (unsigned short*)alloc((size_t)M * Dd * 2);
    unsigned short* vh   = (unsigned short*)alloc((size_t)M * Dd * 2);
    unsigned short* vt   = (unsigned short*)alloc((size_t)M * Dd * 2);  // [B,H,DH,S]
    float* sc            = (float*)alloc((size_t)Bb * Hh * Ss * Ss * 4);
    unsigned short* ph   = (unsigned short*)alloc((size_t)Bb * Hh * Ss * Ss * 2);
    unsigned short* ctxh = (unsigned short*)alloc((size_t)M * Dd * 2);
    float* tmp           = (float*)alloc((size_t)M * Dd * 4);
    unsigned short* ffh  = (unsigned short*)alloc((size_t)M * Ff * 2);

    // ---- launch helpers ----
    auto linear = [&](const unsigned short* X, const Lin& L, int N, int K,
                      float* Yf, unsigned short* Yh, int act) {
        dim3 g(N / 64, M / 64, 1);
        wmma_gemm_bf16<<<g, 128, 0, stream>>>(X, 0, 0, K, L.w, 0, 0, K,
                                              Yf, Yh, 0, 0, N, L.b, 1.0f, K, act, 1);
    };
    auto ln = [&](const float* x, const float* g, const float* b) {
        ln_bf16_kernel<<<M, 256, 0, stream>>>(x, g, b, hh, Dd, 1e-5f);
    };
    auto resadd = [&](float* x) {
        add_kernel<<<(M * Dd + 255) / 256, 256, 0, stream>>>(x, tmp, M * Dd);
    };
    auto attention = [&](const MHA& p, const unsigned short* qsrc,
                         const unsigned short* kvsrc, const unsigned char* kpm,
                         int causal, float* xres) {
        linear(qsrc, p.q, Dd, Dd, nullptr, qh, 0);
        linear(kvsrc, p.k, Dd, Dd, nullptr, kh, 0);
        linear(kvsrc, p.v, Dd, Dd, nullptr, vh, 0);
        {   // scores[z][q][k] = (Q_h . K_h) / sqrt(DH), z = b*H+h
            dim3 g(Ss / 64, Ss / 64, Bb * Hh);
            wmma_gemm_bf16<<<g, 128, 0, stream>>>(
                qh, (ll)Ss * Dd, DH, Dd,
                kh, (ll)Ss * Dd, DH, Dd,
                sc, nullptr, (ll)Hh * Ss * Ss, (ll)Ss * Ss, Ss,
                nullptr, 0.125f, DH, 0, Hh);
        }
        softmax_bf16<<<dim3(Ss, Bb * Hh), 256, 0, stream>>>(sc, ph, kpm, Ss, Hh, causal);
        // V -> [B,H,DH,S] so the P@V GEMM has contiguous K
        transpose_v_kernel<<<(M * Dd + 255) / 256, 256, 0, stream>>>(
            vh, vt, Ss, Dd, Hh, DH, M * Dd);
        {   // ctx[z][q][d] = sum_k P[z][q][k] * Vt[z][d][k]
            dim3 g(DH / 64, Ss / 64, Bb * Hh);
            wmma_gemm_bf16<<<g, 128, 0, stream>>>(
                ph, (ll)Hh * Ss * Ss, (ll)Ss * Ss, Ss,
                vt, (ll)Hh * DH * Ss, (ll)DH * Ss, Ss,
                nullptr, ctxh, (ll)Ss * Dd, DH, Dd,
                nullptr, 1.0f, Ss, 0, Hh);
        }
        linear(ctxh, p.o, Dd, Dd, tmp, nullptr, 0);
        resadd(xres);
    };

    // ================= encoder =================
    embed_kernel<<<M, 256, 0, stream>>>(src_tok, tok_emb, pe, xe, Ss, Dd, SQRT_D);
    for (int l = 0; l < 2; ++l) {
        ln(xe, enc[l].n1g, enc[l].n1b);
        attention(enc[l].attn, hh, hh, skpm, 0, xe);
        ln(xe, enc[l].n2g, enc[l].n2b);
        linear(hh, enc[l].f1, Ff, Dd, nullptr, ffh, 1);   // GELU
        linear(ffh, enc[l].f2, Dd, Ff, tmp, nullptr, 0);
        resadd(xe);
    }
    cvt_f32_bf16<<<(M * Dd + 255) / 256, 256, 0, stream>>>(xe, memh, M * Dd);

    // ================= decoder =================
    embed_kernel<<<M, 256, 0, stream>>>(tgt_tok, tok_emb, pe, xd, Ss, Dd, SQRT_D);
    for (int l = 0; l < 2; ++l) {
        ln(xd, dec[l].n1g, dec[l].n1b);
        attention(dec[l].attn, hh, hh, tkpm, 1, xd);       // causal self-attn
        ln(xd, dec[l].n2g, dec[l].n2b);
        attention(dec[l].xattn, hh, memh, skpm, 0, xd);    // cross-attn
        ln(xd, dec[l].n3g, dec[l].n3b);
        linear(hh, dec[l].f1, Ff, Dd, nullptr, ffh, 1);    // GELU
        linear(ffh, dec[l].f2, Dd, Ff, tmp, nullptr, 0);
        resadd(xd);
    }

    // ================= output projection =================
    cvt_f32_bf16<<<(M * Dd + 255) / 256, 256, 0, stream>>>(xd, hh, M * Dd);
    {
        dim3 g(Vv / 64, M / 64, 1);
        wmma_gemm_bf16<<<g, 128, 0, stream>>>(hh, 0, 0, Dd, out_w, 0, 0, Dd,
                                              (float*)d_out, nullptr, 0, 0, Vv,
                                              out_b, 1.0f, Dd, 0, 1);
    }
}
